// HumanVAttention_76321568850597
// MI455X (gfx1250) — matrix operand order repaired and checked by reading, not compiled, not run on hardware
//
#include <hip/hip_runtime.h>

// ---------------------------------------------------------------------------
// Problem constants (from reference)
// ---------------------------------------------------------------------------
#define B_    2
#define S_    2048
#define H_    2048
#define NH_   16
#define KVH_  4
#define HD_   128
#define WIN_  512
#define EPS_  1e-6f
#define SCALE_ 0.08838834764831845f   // 128^-0.5
#define MROWS_ (B_ * S_)              // 4096
#define TILES_ (S_ / 16)              // 128 q-tiles per head
#define MAXT_  33                     // key tiles covering a 512-window + tile

#define TDM_ENABLE 1                  // Tensor Data Mover staging for A tiles

typedef __bf16 v8bf  __attribute__((ext_vector_type(8)));
typedef __bf16 v16bf __attribute__((ext_vector_type(16)));
typedef float  v8f   __attribute__((ext_vector_type(8)));
typedef unsigned v4u __attribute__((ext_vector_type(4)));
typedef unsigned v8u __attribute__((ext_vector_type(8)));

__device__ inline v8f wmma_bf16(v16bf a, v16bf b, v8f c) {
  return __builtin_amdgcn_wmma_f32_16x16x32_bf16(
      /*neg_a=*/false, a, /*neg_b=*/false, b,
      /*c_mod=*/(short)0, c, /*reuse_a=*/false, /*reuse_b=*/false);
}

// Build a 16-element bf16 fragment from two contiguous 8-element chunks.
__device__ inline v16bf make_frag(const __bf16* p0, const __bf16* p1) {
  v8bf a = *(const v8bf*)p0;
  v8bf b = *(const v8bf*)p1;
  return __builtin_shufflevector(a, b, 0,1,2,3,4,5,6,7,8,9,10,11,12,13,14,15);
}

#if TDM_ENABLE
// LDS byte offset of a generic pointer into shared memory.
__device__ __forceinline__ unsigned lds_off(void* p) {
  return (unsigned)(unsigned long long)(__attribute__((address_space(3))) void*)p;
}

// Post one TDM 2-D tile load: global (row-major, stride_elems 16-bit elements
// per row) -> packed LDS tile of width_elems x height.  D# layout per CDNA5
// CDNA5 ISA (data_size=2B, type=2, count=1).
__device__ __forceinline__ void tdm_load_2d(unsigned lds_addr, const void* gaddr,
                                            unsigned width_elems, unsigned height,
                                            unsigned stride_elems) {
  unsigned long long ga = (unsigned long long)gaddr;
  v4u g0;
  g0[0] = 1u;                                       // count=1, user descriptor
  g0[1] = lds_addr;                                 // lds_addr[31:0]
  g0[2] = (unsigned)(ga & 0xffffffffu);             // global_addr[31:0]
  g0[3] = (unsigned)((ga >> 32) & 0x01ffffffu)      // global_addr[56:32]
          | (2u << 30);                             // type=2 ("image")
  v8u g1;
  g1[0] = 0x00010000u;                              // wg_mask=0, data_size=1 (2B)
  g1[1] = (width_elems & 0xffffu) << 16;            // tensor_dim0[15:0]
  g1[2] = (width_elems >> 16) | ((height & 0xffffu) << 16);   // dim0 hi | dim1 lo
  g1[3] = ((height >> 16) & 0xffffu) | ((width_elems & 0xffffu) << 16); // dim1 hi | tile_dim0
  g1[4] = height & 0xffffu;                         // tile_dim1 (tile_dim2=0)
  g1[5] = stride_elems;                             // tensor_dim0_stride[31:0]
  g1[6] = 0u;                                       // stride hi, dim1_stride lo
  g1[7] = 0u;                                       // dim1_stride hi
  v4u z4 = {0u, 0u, 0u, 0u};                        // groups 2/3 unused (2-D)
  asm volatile("tensor_load_to_lds %0, %1, %2, %3"
               :: "s"(g0), "s"(g1), "s"(z4), "s"(z4)
               : "memory");
}
#endif

// ---------------------------------------------------------------------------
// fp32 -> (hi, lo) bf16 split:  x ~= hi + lo   (bf16x3 GEMM inputs)
// ---------------------------------------------------------------------------
__global__ void split_kernel(const float* __restrict__ x,
                             __bf16* __restrict__ hi, __bf16* __restrict__ lo,
                             size_t n) {
  size_t i = (size_t)blockIdx.x * blockDim.x + threadIdx.x;
  size_t stride = (size_t)gridDim.x * blockDim.x;
  for (; i < n; i += stride) {
    float v = x[i];
    __bf16 h = (__bf16)v;
    hi[i] = h;
    lo[i] = (__bf16)(v - (float)h);
  }
}

// ---------------------------------------------------------------------------
// bf16x3 tiled GEMM:  C[M,N] = A[M,K] @ W[K,N]  (A,W pre-split hi/lo)
// Block: 256 threads = 8 waves; tile 128x128x32; wave tile 32x64.
// A tiles staged by the Tensor Data Mover (double-buffered, TENSORcnt);
// W tiles staged transposed by the waves (TDM has no transpose).
// ---------------------------------------------------------------------------
#define BM 128
#define BN 128
#define BK 32

__global__ __launch_bounds__(256) void gemm_bf16x3_kernel(
    const __bf16* __restrict__ Ah, const __bf16* __restrict__ Al,
    const __bf16* __restrict__ Wh, const __bf16* __restrict__ Wl,
    float* __restrict__ C, int M, int N, int K) {
  __shared__ __bf16 sAh[2][BM][BK];
  __shared__ __bf16 sAl[2][BM][BK];
  __shared__ __bf16 sBh[BN][BK];   // stored transposed: [n][k]
  __shared__ __bf16 sBl[BN][BK];

  const int tid  = threadIdx.x;
  const int wave = tid >> 5;
  const int lane = tid & 31;
  const int bm = blockIdx.y * BM;
  const int bn = blockIdx.x * BN;
  const int wm = (wave >> 1) * 32;   // wave row offset (4 rows of waves)
  const int wn = (wave & 1) * 64;    // wave col offset (2 cols of waves)

  v8f acc[2][4];
#pragma unroll
  for (int i = 0; i < 2; ++i)
#pragma unroll
    for (int j = 0; j < 4; ++j)
#pragma unroll
      for (int e = 0; e < 8; ++e) acc[i][j][e] = 0.f;

  const int arow = tid >> 1;          // 0..127
  const int acol = (tid & 1) * 16;    // 0 or 16
  const int bkr  = tid >> 3;          // 0..31
  const int bnb  = (tid & 7) * 16;    // 0..112

  int buf = 0;
#if TDM_ENABLE
  if (wave == 0) {
    tdm_load_2d(lds_off(&sAh[0][0][0]), Ah + (size_t)bm * K, BK, BM, (unsigned)K);
    tdm_load_2d(lds_off(&sAl[0][0][0]), Al + (size_t)bm * K, BK, BM, (unsigned)K);
  }
#endif

  for (int k0 = 0; k0 < K; k0 += BK) {
    // ---- stage W tile transposed into LDS ----
    {
      const __bf16* g = Wh + (size_t)(k0 + bkr) * N + bn + bnb;
      v8bf b0 = *(const v8bf*)g, b1 = *(const v8bf*)(g + 8);
#pragma unroll
      for (int e = 0; e < 8; ++e) { sBh[bnb + e][bkr] = b0[e]; sBh[bnb + 8 + e][bkr] = b1[e]; }
      g = Wl + (size_t)(k0 + bkr) * N + bn + bnb;
      b0 = *(const v8bf*)g; b1 = *(const v8bf*)(g + 8);
#pragma unroll
      for (int e = 0; e < 8; ++e) { sBl[bnb + e][bkr] = b0[e]; sBl[bnb + 8 + e][bkr] = b1[e]; }
      if (k0 + BK < K) {  // prefetch next W tile (global_prefetch_b8)
        __builtin_prefetch(Wh + (size_t)(k0 + BK + bkr) * N + bn + bnb, 0, 1);
        __builtin_prefetch(Wl + (size_t)(k0 + BK + bkr) * N + bn + bnb, 0, 1);
      }
    }

#if TDM_ENABLE
    if (wave == 0) __builtin_amdgcn_s_wait_tensorcnt(0);   // current A tile done
    __syncthreads();
    if (wave == 0 && k0 + BK < K) {                        // post next A tile
      tdm_load_2d(lds_off(&sAh[buf ^ 1][0][0]),
                  Ah + (size_t)bm * K + (k0 + BK), BK, BM, (unsigned)K);
      tdm_load_2d(lds_off(&sAl[buf ^ 1][0][0]),
                  Al + (size_t)bm * K + (k0 + BK), BK, BM, (unsigned)K);
    }
#else
    {
      const __bf16* g = Ah + (size_t)(bm + arow) * K + k0 + acol;
      *(v8bf*)&sAh[0][arow][acol]     = *(const v8bf*)g;
      *(v8bf*)&sAh[0][arow][acol + 8] = *(const v8bf*)(g + 8);
      g = Al + (size_t)(bm + arow) * K + k0 + acol;
      *(v8bf*)&sAl[0][arow][acol]     = *(const v8bf*)g;
      *(v8bf*)&sAl[0][arow][acol + 8] = *(const v8bf*)(g + 8);
    }
    __syncthreads();
#endif

    // ---- fragments ----
    v16bf afh[2], afl[2], bfh[4], bfl[4];
    {
      const int ar = wm + (lane & 15);
      const int ak = (lane >> 4) * 8;
#pragma unroll
      for (int i = 0; i < 2; ++i) {
        afh[i] = make_frag(&sAh[buf][ar + i * 16][ak], &sAh[buf][ar + i * 16][ak + 16]);
        afl[i] = make_frag(&sAl[buf][ar + i * 16][ak], &sAl[buf][ar + i * 16][ak + 16]);
      }
      const int bc = wn + (lane & 15);
      const int bk = (lane >> 4) * 8;
#pragma unroll
      for (int j = 0; j < 4; ++j) {
        bfh[j] = make_frag(&sBh[bc + j * 16][bk], &sBh[bc + j * 16][bk + 16]);
        bfl[j] = make_frag(&sBl[bc + j * 16][bk], &sBl[bc + j * 16][bk + 16]);
      }
    }

    // ---- bf16x3: hi*hi + hi*lo + lo*hi ----
#pragma unroll
    for (int i = 0; i < 2; ++i)
#pragma unroll
      for (int j = 0; j < 4; ++j) {
        acc[i][j] = wmma_bf16(afh[i], bfh[j], acc[i][j]);
        acc[i][j] = wmma_bf16(afh[i], bfl[j], acc[i][j]);
        acc[i][j] = wmma_bf16(afl[i], bfh[j], acc[i][j]);
      }
    __syncthreads();
#if TDM_ENABLE
    buf ^= 1;
#endif
  }

  // ---- epilogue: C/D layout — lane n = lane&15, rows g + 8*(lane>=16) ----
  const int cn  = lane & 15;
  const int cm0 = (lane >> 4) * 8;
#pragma unroll
  for (int i = 0; i < 2; ++i)
#pragma unroll
    for (int j = 0; j < 4; ++j)
#pragma unroll
      for (int g = 0; g < 8; ++g) {
        int row = bm + wm + i * 16 + cm0 + g;
        int col = bn + wn + j * 16 + cn;
        C[(size_t)row * N + col] = acc[i][j][g];
      }
}

// ---------------------------------------------------------------------------
// Fused per-head RMSNorm + RoPE; writes hi/lo bf16 in [b][h][s][d] layout.
// One block (128 threads) per (b,s,h) head vector.
// ---------------------------------------------------------------------------
__global__ __launch_bounds__(128) void norm_rope_kernel(
    const float* __restrict__ lin, const float* __restrict__ cosb,
    const float* __restrict__ sinb, const float* __restrict__ w,
    __bf16* __restrict__ oh, __bf16* __restrict__ ol, int heads) {
  __shared__ float red[128];
  const int d   = threadIdx.x;
  const int idx = blockIdx.x;
  const int h   = idx % heads;
  const int s   = (idx / heads) % S_;
  const int bb  = idx / (heads * S_);

  const size_t rowbase = ((size_t)bb * S_ + s) * (size_t)(heads * HD_) + (size_t)h * HD_;
  const float x = lin[rowbase + d];
  red[d] = x * x;
  __syncthreads();
  for (int off = 64; off > 0; off >>= 1) {
    if (d < off) red[d] += red[d + off];
    __syncthreads();
  }
  const float r = rsqrtf(red[0] / (float)HD_ + EPS_);

  const float xn = w[d] * x * r;
  const int d2 = (d < 64) ? d + 64 : d - 64;
  const float xn2 = w[d2] * lin[rowbase + d2] * r;
  const float rh = (d < 64) ? -xn2 : xn2;

  const size_t cb = ((size_t)bb * S_ + s) * HD_ + d;
  const float o = xn * cosb[cb] + rh * sinb[cb];

  const size_t ob = ((size_t)(bb * heads + h) * S_ + s) * HD_ + d;
  __bf16 hi = (__bf16)o;
  oh[ob] = hi;
  ol[ob] = (__bf16)(o - (float)hi);
}

// ---------------------------------------------------------------------------
// V -> bf16 transposed to [b][hk][d][s] so P@V B-fragments are contiguous.
// ---------------------------------------------------------------------------
__global__ void vtrans_kernel(const float* __restrict__ vlin, __bf16* __restrict__ Vt) {
  const size_t n = (size_t)B_ * KVH_ * S_ * HD_;
  size_t i = (size_t)blockIdx.x * blockDim.x + threadIdx.x;
  size_t stride = (size_t)gridDim.x * blockDim.x;
  for (; i < n; i += stride) {
    int d = (int)(i % HD_);
    size_t r = i / HD_;
    int s = (int)(r % S_);
    size_t r2 = r / S_;
    int h  = (int)(r2 % KVH_);
    int bb = (int)(r2 / KVH_);
    float v = vlin[((size_t)bb * S_ + s) * (KVH_ * HD_) + (size_t)h * HD_ + d];
    Vt[((size_t)(bb * KVH_ + h) * HD_ + d) * S_ + s] = (__bf16)v;
  }
}

// ---------------------------------------------------------------------------
// Sliding-window attention. Block = 128 threads = 4 waves; each wave owns one
// 16-row query tile. Scores via bf16x3 WMMA, exact 2-pass softmax over the
// LDS-resident band, dense attn band written out, P@V via bf16 WMMA.
// ---------------------------------------------------------------------------
__global__ __launch_bounds__(128) void attn_kernel(
    const __bf16* __restrict__ Qh, const __bf16* __restrict__ Ql,
    const __bf16* __restrict__ Kh, const __bf16* __restrict__ Kl,
    const __bf16* __restrict__ Vt,
    float* __restrict__ attn, float* __restrict__ ctx) {
  extern __shared__ float smem[];                 // 4 waves * MAXT_*256 floats
  __shared__ float redmax[4][16];
  __shared__ float redsum[4][16];

  const int tid  = threadIdx.x;
  const int wave = tid >> 5;
  const int lane = tid & 31;

  const int t  = blockIdx.x * 4 + wave;           // global q-tile id
  const int b  = t / (NH_ * TILES_);
  const int rm = t % (NH_ * TILES_);
  const int h  = rm / TILES_;
  const int tq = rm % TILES_;
  const int hk = h / (NH_ / KVH_);
  const int qs = tq * 16;

  float* ssc = smem + (size_t)wave * (MAXT_ * 256);

  // ---- Q fragments: 4 K-chunks of 32, hi and lo ----
  v16bf qfh[4], qfl[4];
  {
    const int row = qs + (lane & 15);
    const int k0  = (lane >> 4) * 8;
    const size_t base = ((size_t)(b * NH_ + h) * S_ + row) * HD_;
#pragma unroll
    for (int c = 0; c < 4; ++c) {
      const __bf16* p = Qh + base + c * 32 + k0;
      qfh[c] = make_frag(p, p + 16);
      const __bf16* p2 = Ql + base + c * 32 + k0;
      qfl[c] = make_frag(p2, p2 + 16);
    }
  }

  int kt0 = tq - 32; if (kt0 < 0) kt0 = 0;
  const int ntiles = tq - kt0 + 1;                // <= 33
  const size_t kbase0 = (size_t)(b * KVH_ + hk) * S_ * HD_;

  // ---- phase 1: scores S = Q K^T into LDS band ----
  for (int jj = 0; jj < ntiles; ++jj) {
    const int kt = kt0 + jj;
    v8f acc;
#pragma unroll
    for (int e = 0; e < 8; ++e) acc[e] = 0.f;

    const int key = kt * 16 + (lane & 15);
    const size_t kb = kbase0 + (size_t)key * HD_;
    const int kk = (lane >> 4) * 8;
#pragma unroll
    for (int c = 0; c < 4; ++c) {
      const __bf16* ph = Kh + kb + c * 32 + kk;
      v16bf bh = make_frag(ph, ph + 16);
      const __bf16* pl = Kl + kb + c * 32 + kk;
      v16bf bl = make_frag(pl, pl + 16);
      acc = wmma_bf16(qfh[c], bh, acc);
      acc = wmma_bf16(qfh[c], bl, acc);
      acc = wmma_bf16(qfl[c], bh, acc);
    }
    const int n  = lane & 15;
    const int m0 = (lane >> 4) * 8;
    const int j  = kt * 16 + n;
#pragma unroll
    for (int g = 0; g < 8; ++g) {
      const int i = qs + m0 + g;
      float s = acc[g] * SCALE_;
      if (j > i || j < i - WIN_) s = -3.0e38f;    // masked
      ssc[jj * 256 + (m0 + g) * 16 + n] = s;
    }
  }
  __syncthreads();

  // ---- phase 2: per-row max & sum (lanes 0..15 own rows) ----
  if (lane < 16) {
    const int ncols = ntiles * 16;
    float mx = -3.0e38f;
    for (int c = 0; c < ncols; ++c) {
      float s = ssc[(c >> 4) * 256 + lane * 16 + (c & 15)];
      mx = fmaxf(mx, s);
    }
    float sum = 0.f;
    for (int c = 0; c < ncols; ++c) {
      float s = ssc[(c >> 4) * 256 + lane * 16 + (c & 15)];
      sum += __expf(s - mx);
    }
    redmax[wave][lane] = mx;
    redsum[wave][lane] = sum;
  }
  __syncthreads();

  // ---- phase 3: normalize, write dense attn band, keep probs in LDS ----
  for (int jj = 0; jj < ntiles; ++jj) {
    const int kt = kt0 + jj;
    for (int idx = lane; idx < 256; idx += 32) {
      const int m = idx >> 4, n = idx & 15;
      float s = ssc[jj * 256 + idx];
      float p = __expf(s - redmax[wave][m]) / redsum[wave][m];
      ssc[jj * 256 + idx] = p;
      const int i = qs + m, j = kt * 16 + n;
      attn[((size_t)(b * NH_ + h) * S_ + i) * S_ + j] = p;
    }
  }
  __syncthreads();

  // ---- phase 4: ctx = P @ V (bf16 WMMA, 32-key chunks) ----
  v8f oacc[8];
#pragma unroll
  for (int ns = 0; ns < 8; ++ns)
#pragma unroll
    for (int e = 0; e < 8; ++e) oacc[ns][e] = 0.f;

  const int npairs = (ntiles + 1) >> 1;
  const size_t vbase = (size_t)(b * KVH_ + hk) * HD_ * S_;
  for (int p = 0; p < npairs; ++p) {
    // A fragment: probs 16 x 32(keys)
    v16bf pa;
    {
      const int m  = lane & 15;
      const int k0 = (lane >> 4) * 16;
#pragma unroll
      for (int e = 0; e < 16; ++e) {
        const int ke = k0 + e;
        const int jj = p * 2 + (ke >> 4);
        float v = (jj < ntiles) ? ssc[jj * 256 + m * 16 + (ke & 15)] : 0.f;
        pa[e] = (__bf16)v;
      }
    }
    const int kb   = (lane >> 4) * 16;
    const int keyb = (kt0 + p * 2) * 16 + kb;
#pragma unroll
    for (int ns = 0; ns < 8; ++ns) {
      const int dim = ns * 16 + (lane & 15);
      const __bf16* vp = Vt + vbase + (size_t)dim * S_;
      v16bf vb;
#pragma unroll
      for (int e = 0; e < 16; ++e) {
        int key = keyb + e;
        if (key > S_ - 1) key = S_ - 1;           // overhang (prob==0)
        vb[e] = vp[key];
      }
      oacc[ns] = wmma_bf16(pa, vb, oacc[ns]);
    }
  }

  // ---- write ctx [b][s][h*HD+d] fp32 ----
  const int n  = lane & 15;
  const int m0 = (lane >> 4) * 8;
#pragma unroll
  for (int ns = 0; ns < 8; ++ns)
#pragma unroll
    for (int g = 0; g < 8; ++g) {
      const int row = qs + m0 + g;
      const int dim = ns * 16 + n;
      ctx[((size_t)b * S_ + row) * (NH_ * HD_) + (size_t)h * HD_ + dim] = oacc[ns][g];
    }
}

// ---------------------------------------------------------------------------
// Host launcher
// ---------------------------------------------------------------------------
extern "C" void kernel_launch(void* const* d_in, const int* in_sizes, int n_in,
                              void* d_out, int out_size, void* d_ws, size_t ws_size,
                              hipStream_t stream) {
  const float* hidden = (const float*)d_in[0];
  const float* cosb   = (const float*)d_in[1];
  const float* sinb   = (const float*)d_in[2];
  const float* wq     = (const float*)d_in[3];
  const float* wk     = (const float*)d_in[4];
  const float* wv     = (const float*)d_in[5];
  const float* wo     = (const float*)d_in[6];
  const float* qnw    = (const float*)d_in[7];
  const float* knw    = (const float*)d_in[8];

  float* out  = (float*)d_out;                          // [B*S, H]
  float* attn = out + (size_t)B_ * S_ * H_;             // [B, NH, S, S]

  // workspace bump allocator (256B aligned)
  char* ws = (char*)d_ws;
  auto alloc = [&](size_t bytes) -> void* {
    void* p = (void*)ws;
    ws += (bytes + 255) & ~(size_t)255;
    return p;
  };

  const size_t nA  = (size_t)MROWS_ * H_;               // hidden elements
  const size_t nWq = (size_t)H_ * (NH_ * HD_);
  const size_t nWk = (size_t)H_ * (KVH_ * HD_);
  const size_t nWo = (size_t)(NH_ * HD_) * H_;
  const size_t nQ  = (size_t)B_ * NH_ * S_ * HD_;
  const size_t nK  = (size_t)B_ * KVH_ * S_ * HD_;

  __bf16* Ah  = (__bf16*)alloc(nA * 2);
  __bf16* Al  = (__bf16*)alloc(nA * 2);
  __bf16* Wqh = (__bf16*)alloc(nWq * 2);
  __bf16* Wql = (__bf16*)alloc(nWq * 2);
  __bf16* Wkh = (__bf16*)alloc(nWk * 2);
  __bf16* Wkl = (__bf16*)alloc(nWk * 2);
  __bf16* Wvh = (__bf16*)alloc(nWk * 2);
  __bf16* Wvl = (__bf16*)alloc(nWk * 2);
  __bf16* Woh = (__bf16*)alloc(nWo * 2);
  __bf16* Wol = (__bf16*)alloc(nWo * 2);
  float*  qlin = (float*)alloc(nQ * 4);
  float*  klin = (float*)alloc(nK * 4);
  float*  vlin = (float*)alloc(nK * 4);
  __bf16* Qh  = (__bf16*)alloc(nQ * 2);
  __bf16* Ql  = (__bf16*)alloc(nQ * 2);
  __bf16* Kh  = (__bf16*)alloc(nK * 2);
  __bf16* Kl  = (__bf16*)alloc(nK * 2);
  __bf16* Vt  = (__bf16*)alloc(nK * 2);
  float*  ctx = (float*)alloc(nQ * 4);
  __bf16* Ch  = (__bf16*)alloc(nQ * 2);
  __bf16* Cl  = (__bf16*)alloc(nQ * 2);

  // 1) split fp32 -> hi/lo bf16
  split_kernel<<<2048, 256, 0, stream>>>(hidden, Ah, Al, nA);
  split_kernel<<<2048, 256, 0, stream>>>(wq, Wqh, Wql, nWq);
  split_kernel<<<1024, 256, 0, stream>>>(wk, Wkh, Wkl, nWk);
  split_kernel<<<1024, 256, 0, stream>>>(wv, Wvh, Wvl, nWk);
  split_kernel<<<2048, 256, 0, stream>>>(wo, Woh, Wol, nWo);

  // 2) QKV projections (bf16x3 WMMA GEMMs, TDM-staged A tiles)
  gemm_bf16x3_kernel<<<dim3((NH_ * HD_) / BN, MROWS_ / BM), 256, 0, stream>>>(
      Ah, Al, Wqh, Wql, qlin, MROWS_, NH_ * HD_, H_);
  gemm_bf16x3_kernel<<<dim3((KVH_ * HD_) / BN, MROWS_ / BM), 256, 0, stream>>>(
      Ah, Al, Wkh, Wkl, klin, MROWS_, KVH_ * HD_, H_);
  gemm_bf16x3_kernel<<<dim3((KVH_ * HD_) / BN, MROWS_ / BM), 256, 0, stream>>>(
      Ah, Al, Wvh, Wvl, vlin, MROWS_, KVH_ * HD_, H_);

  // 3) RMSNorm + RoPE -> head-major hi/lo bf16
  norm_rope_kernel<<<B_ * S_ * NH_, 128, 0, stream>>>(qlin, cosb, sinb, qnw, Qh, Ql, NH_);
  norm_rope_kernel<<<B_ * S_ * KVH_, 128, 0, stream>>>(klin, cosb, sinb, knw, Kh, Kl, KVH_);

  // 4) V -> bf16, transposed [b][hk][d][s]
  vtrans_kernel<<<2048, 256, 0, stream>>>(vlin, Vt);

  // 5) dense attn tensor: zero outside the sliding-window band
  hipMemsetAsync(attn, 0, (size_t)B_ * NH_ * S_ * S_ * sizeof(float), stream);

  // 6) windowed attention (writes attn band + ctx)
  const size_t attn_lds = (size_t)4 * MAXT_ * 256 * sizeof(float);  // 135 KB
  attn_kernel<<<(B_ * NH_ * TILES_) / 4, 128, attn_lds, stream>>>(
      Qh, Ql, Kh, Kl, Vt, attn, ctx);

  // 7) output projection (bf16x3 WMMA GEMM)
  split_kernel<<<2048, 256, 0, stream>>>(ctx, Ch, Cl, nQ);
  gemm_bf16x3_kernel<<<dim3(H_ / BN, MROWS_ / BM), 256, 0, stream>>>(
      Ch, Cl, Woh, Wol, out, MROWS_, H_, NH_ * HD_);
}
